// TestFusedARRMSNormNoQuantModel_28664611733561
// MI455X (gfx1250) — matrix operand order repaired and checked by reading, not compiled, not run on hardware
//
#include <hip/hip_runtime.h>
#include <hip/hip_bf16.h>

// TestFusedARRMSNormNoQuantModel: relu -> rmsnorm -> GEMM -> add+rmsnorm -> GEMM -> add+rmsnorm
// T=16384, H=4096, all fp32. GEMMs run as bf16x3 split (hi/lo) on v_wmma_f32_16x16x32_bf16,
// f32 accumulation => ~1e-5 rel err at 3/8 the WMMA issue cost of the native f32 16x16x4 path.
//
// v2: hi/lo split hoisted into the RMSNorm producers (GEMM hot loop is now pure
//     data movement + WMMA), wave tile 64x64 (48 WMMA per K-chunk, 1.5 WMMA per
//     ds_load_b128), software-pipelined global->LDS staging.
//
// Workspace (~670 MB):
//   Yh/Yl : T*H bf16 (normalized activations, hi/lo, GEMM A operand)
//   R     : T*H f32  (residual chain)
//   W0h/W0l/W1h/W1l : H*H bf16 (transposed, K-contiguous, hi/lo)

#define T_DIM 16384
#define H_DIM 4096
#define EPSV  1e-5f

#define BM 256
#define BN 128
#define BK 32
#define LDT 40   // padded LDS row stride in ushorts (80 B) - 16B aligned, spreads banks

typedef __attribute__((ext_vector_type(16))) __bf16          v16bf;
typedef __attribute__((ext_vector_type(8)))  float           v8f;
typedef __attribute__((ext_vector_type(8)))  unsigned short  us8;
typedef __attribute__((ext_vector_type(16))) unsigned short  us16;
typedef __attribute__((ext_vector_type(4)))  unsigned short  us4;

static __device__ __forceinline__ unsigned short f2bf(float f) {
  unsigned int u = __float_as_uint(f);
  u += 0x7FFFu + ((u >> 16) & 1u);          // round-to-nearest-even
  return (unsigned short)(u >> 16);
}
static __device__ __forceinline__ float bf2f(unsigned short h) {
  return __uint_as_float(((unsigned int)h) << 16);
}
static __device__ __forceinline__ v16bf mkfrag(us8 lo8, us8 hi8) {
  union { us8 h[2]; us16 w; } u;
  u.h[0] = lo8; u.h[1] = hi8;
  return __builtin_bit_cast(v16bf, u.w);
}

// ---------------------------------------------------------------------------
// Weight prep: transpose W (H x H row-major) into K-contiguous wT[n][k],
// split into bf16 hi/lo. 32x32 LDS tile transpose, coalesced both directions.
// ---------------------------------------------------------------------------
__global__ __launch_bounds__(256)
void wsplit_kernel(const float* __restrict__ W,
                   unsigned short* __restrict__ Th,
                   unsigned short* __restrict__ Tl) {
  __shared__ float tile[32][33];
  const int gx = blockIdx.x * 32;   // n-tile
  const int gy = blockIdx.y * 32;   // k-tile
  const int tx = threadIdx.x;       // 0..31
  const int ty = threadIdx.y;       // 0..7
#pragma unroll
  for (int i = 0; i < 4; ++i) {
    int r = ty + i * 8;
    tile[r][tx] = W[(size_t)(gy + r) * H_DIM + gx + tx];
  }
  __syncthreads();
#pragma unroll
  for (int i = 0; i < 4; ++i) {
    int r = ty + i * 8;                       // local n
    float v = tile[tx][r];                    // = W[gy+tx][gx+r]
    unsigned short h = f2bf(v);
    unsigned short l = f2bf(v - bf2f(h));
    size_t o = (size_t)(gx + r) * H_DIM + gy + tx;
    Th[o] = h;
    Tl[o] = l;
  }
}

// ---------------------------------------------------------------------------
// Row RMSNorm (one 256-thread block per row, 8 wave32s), optional input ReLU.
// Emits either bf16 hi/lo split (GEMM operand) or f32 (final output).
// ---------------------------------------------------------------------------
__global__ __launch_bounds__(256)
void rmsnorm_kernel(const float* __restrict__ in, const float* __restrict__ g,
                    unsigned short* __restrict__ outh,
                    unsigned short* __restrict__ outl,
                    float* __restrict__ outf, int do_relu) {
  const size_t row = blockIdx.x;
  const float* rp = in + row * H_DIM;

  float z[16];
  float ss = 0.f;
#pragma unroll
  for (int i = 0; i < 4; ++i) {
    int c = ((int)threadIdx.x + i * 256) * 4;
    float4 v = *(const float4*)(rp + c);
    if (do_relu) {
      v.x = fmaxf(v.x, 0.f); v.y = fmaxf(v.y, 0.f);
      v.z = fmaxf(v.z, 0.f); v.w = fmaxf(v.w, 0.f);
    }
    z[i * 4 + 0] = v.x; z[i * 4 + 1] = v.y; z[i * 4 + 2] = v.z; z[i * 4 + 3] = v.w;
    ss += v.x * v.x + v.y * v.y + v.z * v.z + v.w * v.w;
  }
  // wave32 reduce, then cross-wave via LDS
#pragma unroll
  for (int off = 16; off > 0; off >>= 1) ss += __shfl_xor(ss, off, 32);
  __shared__ float red[8];
  const int wid = threadIdx.x >> 5, lane = threadIdx.x & 31;
  if (lane == 0) red[wid] = ss;
  __syncthreads();
  if (threadIdx.x < 32) {
    float v = (lane < 8) ? red[lane] : 0.f;
#pragma unroll
    for (int off = 4; off > 0; off >>= 1) v += __shfl_xor(v, off, 32);
    if (lane == 0) red[0] = v;
  }
  __syncthreads();
  const float inv = rsqrtf(red[0] * (1.0f / H_DIM) + EPSV);

#pragma unroll
  for (int i = 0; i < 4; ++i) {
    int c = ((int)threadIdx.x + i * 256) * 4;
    float4 gv = *(const float4*)(g + c);
    float ov[4];
    ov[0] = z[i * 4 + 0] * inv * gv.x;
    ov[1] = z[i * 4 + 1] * inv * gv.y;
    ov[2] = z[i * 4 + 2] * inv * gv.z;
    ov[3] = z[i * 4 + 3] * inv * gv.w;
    if (outf) {
      float4 o; o.x = ov[0]; o.y = ov[1]; o.z = ov[2]; o.w = ov[3];
      *(float4*)(outf + row * H_DIM + c) = o;
    } else {
      us4 h, l;
#pragma unroll
      for (int j = 0; j < 4; ++j) {
        unsigned short hh = f2bf(ov[j]);
        h[j] = hh;
        l[j] = f2bf(ov[j] - bf2f(hh));
      }
      *(us4*)(outh + row * H_DIM + c) = h;
      *(us4*)(outl + row * H_DIM + c) = l;
    }
  }
}

// ---------------------------------------------------------------------------
// GEMM: out[m][n] = sum_k A[m][k]*W[k][n] + addend[m][n]  (addend opt. ReLU'd)
// A and W pre-split bf16 hi/lo, W pre-transposed (K-contiguous rows).
// WG 256 thr (8 waves), tile 256x128, BK=32; wave tile 64x64 -> 4x4 acc tiles,
// 48 WMMAs per K-chunk. Global->LDS staging software-pipelined one chunk ahead.
// ---------------------------------------------------------------------------
__global__ __launch_bounds__(256)
void gemm_bf16x3_kernel(const unsigned short* __restrict__ Ahg,
                        const unsigned short* __restrict__ Alg,
                        const unsigned short* __restrict__ Whg,
                        const unsigned short* __restrict__ Wlg,
                        const float* __restrict__ addend,
                        float* __restrict__ out,
                        int relu_addend) {
  __shared__ unsigned short Ah[BM][LDT];
  __shared__ unsigned short Al[BM][LDT];
  __shared__ unsigned short Bh[BN][LDT];
  __shared__ unsigned short Bl[BN][LDT];

  const int tid  = threadIdx.x;
  const int n0   = blockIdx.x * BN;
  const int m0   = blockIdx.y * BM;
  const int wave = tid >> 5;
  const int lane = tid & 31;
  const int wm   = wave & 3;     // -> M offset wm*64
  const int wn   = wave >> 2;    // -> N offset wn*64
  const int hl   = lane >> 4;    // half-wave select
  const int lr   = lane & 15;

  const v8f zacc = {0.f, 0.f, 0.f, 0.f, 0.f, 0.f, 0.f, 0.f};
  v8f acc[4][4];
#pragma unroll
  for (int mi = 0; mi < 4; ++mi)
#pragma unroll
    for (int ni = 0; ni < 4; ++ni) acc[mi][ni] = zacc;

  // staged registers for next K-chunk (12 x us8 = 192 B/thread)
  us8 rAh[4], rAl[4], rBh[2], rBl[2];

  auto load_tile = [&](int k0) {
#pragma unroll
    for (int i = 0; i < 4; ++i) {
      int idx = tid + i * 256;               // 0..1023 us8 slots (256 rows x 4)
      int r = idx >> 2, c8 = idx & 3;
      size_t o = (size_t)(m0 + r) * H_DIM + k0 + c8 * 8;
      rAh[i] = *(const us8*)(Ahg + o);
      rAl[i] = *(const us8*)(Alg + o);
    }
#pragma unroll
    for (int i = 0; i < 2; ++i) {
      int idx = tid + i * 256;               // 0..511 us8 slots (128 rows x 4)
      int r = idx >> 2, c8 = idx & 3;
      size_t o = (size_t)(n0 + r) * H_DIM + k0 + c8 * 8;
      rBh[i] = *(const us8*)(Whg + o);
      rBl[i] = *(const us8*)(Wlg + o);
    }
  };

  load_tile(0);

  for (int k0 = 0; k0 < H_DIM; k0 += BK) {
    // ---- commit staged chunk to LDS ----
#pragma unroll
    for (int i = 0; i < 4; ++i) {
      int idx = tid + i * 256;
      int r = idx >> 2, c8 = idx & 3;
      *(us8*)&Ah[r][c8 * 8] = rAh[i];
      *(us8*)&Al[r][c8 * 8] = rAl[i];
    }
#pragma unroll
    for (int i = 0; i < 2; ++i) {
      int idx = tid + i * 256;
      int r = idx >> 2, c8 = idx & 3;
      *(us8*)&Bh[r][c8 * 8] = rBh[i];
      *(us8*)&Bl[r][c8 * 8] = rBl[i];
    }
    __syncthreads();

    // ---- issue next chunk's global loads; they fly under the WMMAs below ----
    if (k0 + BK < H_DIM) load_tile(k0 + BK);

    // ---- fragments (ISA 16-bit operand layouts) ----
    // A 16x32: lanes 0-15 (M=lr) hold K {0..7,16..23}; lanes 16-31 K {8..15,24..31}
    // B 32x16: lanes 0-15 (N=lr) hold K {0..7,16..23}; lanes 16-31 K {8..15,24..31}
    v16bf afh[4], afl[4];
#pragma unroll
    for (int mi = 0; mi < 4; ++mi) {
      int am = wm * 64 + mi * 16 + lr;
      afh[mi] = mkfrag(*(const us8*)&Ah[am][hl * 8], *(const us8*)&Ah[am][hl * 8 + 16]);
      afl[mi] = mkfrag(*(const us8*)&Al[am][hl * 8], *(const us8*)&Al[am][hl * 8 + 16]);
    }
#pragma unroll
    for (int ni = 0; ni < 4; ++ni) {
      int bn = wn * 64 + ni * 16 + lr;
      v16bf bh = mkfrag(*(const us8*)&Bh[bn][hl * 8], *(const us8*)&Bh[bn][hl * 8 + 16]);
      v16bf bl = mkfrag(*(const us8*)&Bl[bn][hl * 8], *(const us8*)&Bl[bn][hl * 8 + 16]);
      // bf16x3: hi*hi + hi*lo + lo*hi, f32 accumulate
#pragma unroll
      for (int mi = 0; mi < 4; ++mi) {
        v8f c = acc[mi][ni];
        c = __builtin_amdgcn_wmma_f32_16x16x32_bf16(false, afh[mi], false, bh,
                                                    (short)0, c, false, false);
        c = __builtin_amdgcn_wmma_f32_16x16x32_bf16(false, afh[mi], false, bl,
                                                    (short)0, c, false, false);
        c = __builtin_amdgcn_wmma_f32_16x16x32_bf16(false, afl[mi], false, bh,
                                                    (short)0, c, false, false);
        acc[mi][ni] = c;
      }
    }
    __syncthreads();
  }

  // ---- epilogue: fused residual add (optional ReLU on addend) ----
  // C/D layout: VGPR j -> M=j (lanes 0-15) / M=j+8 (lanes 16-31), N=lr
#pragma unroll
  for (int mi = 0; mi < 4; ++mi)
#pragma unroll
    for (int ni = 0; ni < 4; ++ni) {
#pragma unroll
      for (int j = 0; j < 8; ++j) {
        int m = m0 + wm * 64 + mi * 16 + hl * 8 + j;
        int n = n0 + wn * 64 + ni * 16 + lr;
        size_t o = (size_t)m * H_DIM + n;
        float add = addend[o];
        if (relu_addend) add = fmaxf(add, 0.f);
        out[o] = acc[mi][ni][j] + add;
      }
    }
}

// ---------------------------------------------------------------------------
extern "C" void kernel_launch(void* const* d_in, const int* in_sizes, int n_in,
                              void* d_out, int out_size, void* d_ws, size_t ws_size,
                              hipStream_t stream) {
  (void)in_sizes; (void)n_in; (void)out_size; (void)ws_size;
  const float* x  = (const float*)d_in[0];
  const float* g0 = (const float*)d_in[1];
  const float* g1 = (const float*)d_in[2];
  const float* g2 = (const float*)d_in[3];
  const float* w0 = (const float*)d_in[4];
  const float* w1 = (const float*)d_in[5];
  float* out = (float*)d_out;

  const size_t szTH = (size_t)T_DIM * H_DIM;
  const size_t szHH = (size_t)H_DIM * H_DIM;
  char* ws = (char*)d_ws;
  unsigned short* Yh = (unsigned short*)ws;  ws += szTH * sizeof(unsigned short);
  unsigned short* Yl = (unsigned short*)ws;  ws += szTH * sizeof(unsigned short);
  float* R = (float*)ws;                     ws += szTH * sizeof(float);
  unsigned short* W0h = (unsigned short*)ws; ws += szHH * sizeof(unsigned short);
  unsigned short* W0l = (unsigned short*)ws; ws += szHH * sizeof(unsigned short);
  unsigned short* W1h = (unsigned short*)ws; ws += szHH * sizeof(unsigned short);
  unsigned short* W1l = (unsigned short*)ws;

  dim3 tgrid(H_DIM / 32, H_DIM / 32), tblk(32, 8);
  wsplit_kernel<<<tgrid, tblk, 0, stream>>>(w0, W0h, W0l);
  wsplit_kernel<<<tgrid, tblk, 0, stream>>>(w1, W1h, W1l);

  dim3 ggrid(H_DIM / BN, T_DIM / BM);

  // z = relu(x); (Yh,Yl) = split(rmsnorm(z, g0))
  rmsnorm_kernel<<<T_DIM, 256, 0, stream>>>(x, g0, Yh, Yl, nullptr, 1);
  // R = Y @ w0 + relu(x)
  gemm_bf16x3_kernel<<<ggrid, 256, 0, stream>>>(Yh, Yl, W0h, W0l, x, R, 1);
  // (Yh,Yl) = split(rmsnorm(R, g1))
  rmsnorm_kernel<<<T_DIM, 256, 0, stream>>>(R, g1, Yh, Yl, nullptr, 0);
  // R = Y @ w1 + R   (in-place per-element read-then-write by owning lane)
  gemm_bf16x3_kernel<<<ggrid, 256, 0, stream>>>(Yh, Yl, W1h, W1l, R, R, 0);
  // out = rmsnorm(R, g2)
  rmsnorm_kernel<<<T_DIM, 256, 0, stream>>>(R, g2, nullptr, nullptr, out, 0);
}
